// GNNEncoder_14628658610450
// MI455X (gfx1250) — compile-verified, matrix operand order checked by reading, not run on hardware
//
#include <hip/hip_runtime.h>
#include <hip/hip_bf16.h>
#include <stdint.h>

#define NN 100000
#define EE 800000
#define NT_ 4
#define D_ET_ 32
#define D_EA_ 32
#define EDGE_DIM_ 16

typedef __attribute__((ext_vector_type(16))) _Float16 v16h;
typedef __attribute__((ext_vector_type(8)))  _Float16 v8h;
typedef __attribute__((ext_vector_type(8)))  float    v8f;

__device__ __forceinline__ float lrelu_f(float v){ return v > 0.f ? v : 0.2f*v; }

// ---- CDNA5 async copy: global -> LDS, 16B per lane, tracked by ASYNCcnt ----
__device__ __forceinline__ void async_ld16(unsigned lds_byte_off, const void* gptr){
    asm volatile("global_load_async_to_lds_b128 %0, %1, off"
                 :: "v"(lds_byte_off), "v"(gptr) : "memory");
}
__device__ __forceinline__ void wait_async0(){
    asm volatile("s_wait_asynccnt 0x0" ::: "memory");
}

// ---- WMMA fragment helpers (16-bit A layout per CDNA5 ISA §7.12.2) ----
// lanes 0-15: row M = lane&15, K = {kh..kh+7, 16+kh..16+kh+7}, kh = 8*(lane>>4)
__device__ __forceinline__ v16h load_frag16(const _Float16* __restrict__ base,
                                            int row0, int k0, int ld, int lane){
    int r  = row0 + (lane & 15);
    int kh = (lane >> 4) << 3;
    const _Float16* p = base + (size_t)r * ld + (size_t)(k0 + kh);
    v8h lo = *(const v8h*)(p);
    v8h hi = *(const v8h*)(p + 16);
    v16h out;
#pragma unroll
    for (int i = 0; i < 8; ++i){ out[i] = lo[i]; out[i+8] = hi[i]; }
    return out;
}

// same layout, but from an LDS-resident tile (emits ds_load_b128)
__device__ __forceinline__ v16h load_frag_lds(const _Float16* sm, int k0, int ld, int lane){
    int r  = lane & 15;
    int kh = (lane >> 4) << 3;
    const _Float16* p = sm + r * ld + (k0 + kh);
    v8h lo = *(const v8h*)(p);
    v8h hi = *(const v8h*)(p + 16);
    v16h out;
#pragma unroll
    for (int i = 0; i < 8; ++i){ out[i] = lo[i]; out[i+8] = hi[i]; }
    return out;
}

__device__ __forceinline__ v8f wmma_f16(v16h a, v16h b, v8f c){
    return __builtin_amdgcn_wmma_f32_16x16x32_f16(false, a, false, b, (short)0, c, false, false);
}

// ---- ordered-int float encoding for segment max via atomicMax ----
__device__ __forceinline__ unsigned fenc(float x){
    unsigned u = __float_as_uint(x);
    return (u & 0x80000000u) ? ~u : (u | 0x80000000u);
}
__device__ __forceinline__ float fdec(unsigned m){
    unsigned u = (m & 0x80000000u) ? (m & 0x7FFFFFFFu) : ~m;
    return __uint_as_float(u);
}

// ------------------------- elementwise / prep kernels -------------------------
__global__ void k_f32_to_f16(const float* __restrict__ in, _Float16* __restrict__ out, size_t n){
    size_t i = (size_t)blockIdx.x * 256 + threadIdx.x;
    if (i < n) out[i] = (_Float16)in[i];
}

// in: [B][K][Ncols] f32 -> out: [B][Ncols][K] f16 (so WMMA B-fragments read contiguous K)
__global__ void k_transpose_f16(const float* __restrict__ in, _Float16* __restrict__ out,
                                int K, int Ncols, size_t total){
    size_t idx = (size_t)blockIdx.x * 256 + threadIdx.x;
    if (idx >= total) return;
    int n = (int)(idx % (size_t)Ncols);
    size_t r = idx / (size_t)Ncols;
    int k = (int)(r % (size_t)K);
    size_t t = r / (size_t)K;
    out[(t * (size_t)Ncols + n) * (size_t)K + k] = (_Float16)in[idx];
}

__global__ void k_zero(float* __restrict__ p, size_t n){
    size_t i = (size_t)blockIdx.x * 256 + threadIdx.x;
    if (i < n) p[i] = 0.f;
}

__global__ void k_init_node(unsigned* __restrict__ menc, float* __restrict__ denom, int n){
    int i = blockIdx.x * 256 + threadIdx.x;
    if (i < n){ menc[i] = 0u; denom[i] = 0.f; }
}

// ------------- HeteroLinear: xl = select_t(h @ hw[t] + hb[t]) -------------
// 8 waves/block share one 16-row A tile (async-staged in LDS); each wave owns a
// 16-col slice. 4 accumulators (one per node type), per-row select in epilogue.
__global__ void __launch_bounds__(256) k_hetero_gemm(
    const _Float16* __restrict__ hf16,      // [N][ci] f16
    const _Float16* __restrict__ hwT,       // [NT][co][ci] f16 (transposed)
    const float* __restrict__ hb,           // [NT][co]
    const int* __restrict__ ntype,          // [N]
    float* __restrict__ xl, _Float16* __restrict__ xlf16, int ci, int co)
{
    __shared__ __align__(16) _Float16 smA[16 * 256];   // 8 KB max
    int tid = threadIdx.x;
    int wave = tid >> 5, lane = tid & 31;
    int groups = (co + 127) >> 7;
    int bm = blockIdx.x / groups;
    int bg = blockIdx.x - bm * groups;
    int row0 = bm << 4;

    // stage A rows [row0, row0+16) x ci into LDS via async copies (16B/lane)
    int cpr = ci >> 3;                 // 16B chunks per row
    int chunks = cpr << 4;             // 16 rows
    for (int idx = tid; idx < chunks; idx += 256){
        int r = idx / cpr, c8 = idx - r * cpr;
        async_ld16((unsigned)(size_t)&smA[idx * 8],
                   hf16 + (size_t)(row0 + r) * ci + c8 * 8);
    }
    wait_async0();
    __syncthreads();

    int c0 = (bg << 7) + (wave << 4);
    if (c0 >= co) return;

    v8f acc[NT_] = {};
    for (int k0 = 0; k0 < ci; k0 += 32){
        v16h a = load_frag_lds(smA, k0, ci, lane);
#pragma unroll
        for (int t = 0; t < NT_; ++t){
            v16h b = load_frag16(hwT + (size_t)t * co * ci, c0, k0, ci, lane);
            acc[t] = wmma_f16(a, b, acc[t]);
        }
    }
    int col = c0 + (lane & 15);
    int rbase = row0 + ((lane >> 4) << 3);
#pragma unroll
    for (int v = 0; v < 8; ++v){
        int row = rbase + v;
        int t = ntype[row];
        float val = (t == 0) ? acc[0][v] : (t == 1) ? acc[1][v] : (t == 2) ? acc[2][v] : acc[3][v];
        val += hb[t * co + col];
        xl[(size_t)row * co + col] = val;
        xlf16[(size_t)row * co + col] = (_Float16)val;
    }
}

// ------------------------- per-node attention dot products -------------------------
__global__ void k_node_dots(const float* __restrict__ xl, const float* __restrict__ att,
                            float* __restrict__ s_i, float* __restrict__ s_j, int co){
    int n = blockIdx.x * 256 + threadIdx.x;
    if (n >= NN) return;
    const float* row = xl + (size_t)n * co;
    float si = 0.f, sj = 0.f;
    for (int c = 0; c < co; ++c){ float v = row[c]; si += v * att[c]; sj += v * att[co + c]; }
    s_i[n] = si; s_j[n] = sj;
}

// ------------------------- edge-attr embedding: ea = lrelu(edge_attr @ eaw) -------------------------
__global__ void k_edge_emb(const float* __restrict__ edge_attr, const float* __restrict__ eaw,
                           const float* __restrict__ a4, _Float16* __restrict__ eaf16,
                           float* __restrict__ eadot){
    int e = blockIdx.x * 256 + threadIdx.x;
    if (e >= EE) return;
    float in[EDGE_DIM_];
#pragma unroll
    for (int i = 0; i < EDGE_DIM_; ++i) in[i] = edge_attr[(size_t)e * EDGE_DIM_ + i];
    float dot = 0.f;
#pragma unroll
    for (int c = 0; c < D_EA_; ++c){
        float v = 0.f;
#pragma unroll
        for (int i = 0; i < EDGE_DIM_; ++i) v += in[i] * eaw[i * D_EA_ + c];
        v = lrelu_f(v);
        eaf16[(size_t)e * D_EA_ + c] = (_Float16)v;
        dot += v * a4[c];
    }
    eadot[e] = dot;
}

// ------------------------- attention logits + segment max -------------------------
__global__ void k_alpha(const int* __restrict__ src, const int* __restrict__ dst,
                        const int* __restrict__ etype, const float* __restrict__ s_i,
                        const float* __restrict__ s_j, const float* __restrict__ ete,
                        const float* __restrict__ a3, const float* __restrict__ eadot,
                        float* __restrict__ alphab, unsigned* __restrict__ menc){
    int e = blockIdx.x * 256 + threadIdx.x;
    if (e >= EE) return;
    const float* er = ete + (size_t)etype[e] * D_ET_;
    float etd = 0.f;
#pragma unroll
    for (int d = 0; d < D_ET_; ++d) etd += lrelu_f(er[d]) * a3[d];
    float al = lrelu_f(s_i[dst[e]] + s_j[src[e]] + etd + eadot[e]);
    alphab[e] = al;
    atomicMax(&menc[dst[e]], fenc(al));
}

// ------------------------- exp + segment sum -------------------------
__global__ void k_exp(const int* __restrict__ dst, const float* __restrict__ alphab,
                      const unsigned* __restrict__ menc, float* __restrict__ exv,
                      float* __restrict__ denom){
    int e = blockIdx.x * 256 + threadIdx.x;
    if (e >= EE) return;
    int d = dst[e];
    float ex = __expf(alphab[e] - fdec(menc[d]));
    exv[e] = ex;
    atomicAdd(&denom[d], ex);
}

// ------------- node-side message projection: msgnode = xl @ lw[:co] + lb -------------
__global__ void __launch_bounds__(256) k_gemm_bias(
    const _Float16* __restrict__ A,   // [N][K] f16
    const _Float16* __restrict__ BT,  // [Ncols][ldbt] f16, uses k in [0,K)
    const float* __restrict__ bias, float* __restrict__ C, int K, int Ncols, int ldbt)
{
    __shared__ __align__(16) _Float16 smA[16 * 256];
    int tid = threadIdx.x;
    int wave = tid >> 5, lane = tid & 31;
    int groups = (Ncols + 127) >> 7;
    int bm = blockIdx.x / groups;
    int bg = blockIdx.x - bm * groups;
    int row0 = bm << 4;

    int cpr = K >> 3;
    int chunks = cpr << 4;
    for (int idx = tid; idx < chunks; idx += 256){
        int r = idx / cpr, c8 = idx - r * cpr;
        async_ld16((unsigned)(size_t)&smA[idx * 8],
                   A + (size_t)(row0 + r) * K + c8 * 8);
    }
    wait_async0();
    __syncthreads();

    int c0 = (bg << 7) + (wave << 4);
    if (c0 >= Ncols) return;

    v8f acc = {};
    for (int k0 = 0; k0 < K; k0 += 32){
        v16h a = load_frag_lds(smA, k0, K, lane);
        v16h b = load_frag16(BT, c0, k0, ldbt, lane);
        acc = wmma_f16(a, b, acc);
    }
    int col = c0 + (lane & 15);
    int rbase = row0 + ((lane >> 4) << 3);
    float bia = bias[col];
#pragma unroll
    for (int v = 0; v < 8; ++v)
        C[(size_t)(rbase + v) * Ncols + col] = acc[v] + bia;
}

// ------------- fused edge GEMM + softmax weight + scatter-add -------------
// msg[e] = (msgnode[src[e]] + ea[e] @ lw[co:]) * (exv[e]/(denom[dst]+1e-16)); out[dst] += msg
__global__ void __launch_bounds__(256) k_msg_scatter(
    const _Float16* __restrict__ eaf16,   // [E][32] f16
    const _Float16* __restrict__ lwT,     // [co][co+32] f16, k in [co, co+32)
    const float* __restrict__ msgnode,    // [N][co]
    const int* __restrict__ src, const int* __restrict__ dst,
    const float* __restrict__ exv, const float* __restrict__ denom,
    float* __restrict__ outacc, int co)
{
    __shared__ __align__(16) _Float16 smE[16 * 32];    // 1 KB edge tile
    int tid = threadIdx.x;
    int wave = tid >> 5, lane = tid & 31;
    int groups = (co + 127) >> 7;
    int be = blockIdx.x / groups;
    int bg = blockIdx.x - be * groups;
    int e0 = be << 4;

    // 16 edges x 32 halves = 64 x 16B chunks, contiguous in global
    if (tid < 64)
        async_ld16((unsigned)(size_t)&smE[tid * 8],
                   eaf16 + (size_t)e0 * D_EA_ + tid * 8);
    wait_async0();
    __syncthreads();

    int c0 = (bg << 7) + (wave << 4);
    if (c0 >= co) return;

    v16h a = load_frag_lds(smE, 0, D_EA_, lane);
    v16h b = load_frag16(lwT, c0, co, co + D_EA_, lane);
    v8f acc = {};
    acc = wmma_f16(a, b, acc);

    int col = c0 + (lane & 15);
    int ebase = e0 + ((lane >> 4) << 3);
#pragma unroll
    for (int v = 0; v < 8; ++v){
        int e = ebase + v;
        int s = src[e], d = dst[e];
        float w = exv[e] / (denom[d] + 1e-16f);
        float val = (acc[v] + msgnode[(size_t)s * co + col]) * w;
        atomicAdd(&outacc[(size_t)d * co + col], val);
    }
}

// ------------------------- LayerNorm (+optional ReLU), one wave32 per node -------------------------
__global__ void k_layernorm(const float* __restrict__ accp, const float* __restrict__ g,
                            const float* __restrict__ bb, float* __restrict__ out,
                            int co, int do_relu)
{
    int node = blockIdx.x * (blockDim.x >> 5) + (threadIdx.x >> 5);
    int lane = threadIdx.x & 31;
    if (node >= NN) return;
    int per = co >> 5;
    const float* row = accp + (size_t)node * co;
    float v[8];
    float s = 0.f;
#pragma unroll
    for (int i = 0; i < 8; ++i){ v[i] = 0.f; if (i < per){ v[i] = row[i * 32 + lane]; s += v[i]; } }
    for (int off = 16; off > 0; off >>= 1) s += __shfl_xor(s, off, 32);
    float mu = s / (float)co;
    float var = 0.f;
#pragma unroll
    for (int i = 0; i < 8; ++i){ if (i < per){ float d = v[i] - mu; var += d * d; } }
    for (int off = 16; off > 0; off >>= 1) var += __shfl_xor(var, off, 32);
    float rs = rsqrtf(var / (float)co + 1e-5f);
#pragma unroll
    for (int i = 0; i < 8; ++i){ if (i < per){
        int c = i * 32 + lane;
        float y = (v[i] - mu) * rs * g[c] + bb[c];
        if (do_relu) y = fmaxf(y, 0.f);
        out[(size_t)node * co + c] = y;
    }}
}

// ============================ host orchestration ============================
extern "C" void kernel_launch(void* const* d_in, const int* in_sizes, int n_in,
                              void* d_out, int out_size, void* d_ws, size_t ws_size,
                              hipStream_t stream)
{
    (void)in_sizes; (void)n_in; (void)out_size; (void)ws_size;
    static const int DIMS_[6] = {64, 128, 256, 192, 128, 64};

    const float* x         = (const float*)d_in[0];
    const float* edge_attr = (const float*)d_in[1];
    const int* edge_index  = (const int*)d_in[2 + 5 * 9 + 0];
    const int* node_type   = (const int*)d_in[2 + 5 * 9 + 1];
    const int* edge_type   = (const int*)d_in[2 + 5 * 9 + 2];
    const int* srcp = edge_index;
    const int* dstp = edge_index + EE;

    char* ws = (char*)d_ws;
    size_t off = 0;
    auto carve = [&](size_t bytes) -> char* {
        char* p = ws + off;
        off += (bytes + 255) & ~(size_t)255;
        return p;
    };
    float*    h_cur   = (float*)carve((size_t)NN * 256 * 4);
    float*    hnxt    = (float*)carve((size_t)NN * 256 * 4);
    float*    xl      = (float*)carve((size_t)NN * 256 * 4);
    float*    msgnode = (float*)carve((size_t)NN * 256 * 4);
    _Float16* xlf16   = (_Float16*)carve((size_t)NN * 256 * 2);
    _Float16* hf16    = (_Float16*)carve((size_t)NN * 256 * 2);
    _Float16* eaf16   = (_Float16*)carve((size_t)EE * 32 * 2);
    float*    eadot   = (float*)carve((size_t)EE * 4);
    float*    alphab  = (float*)carve((size_t)EE * 4);
    float*    exv     = (float*)carve((size_t)EE * 4);
    float*    s_i     = (float*)carve((size_t)NN * 4);
    float*    s_j     = (float*)carve((size_t)NN * 4);
    unsigned* menc    = (unsigned*)carve((size_t)NN * 4);
    float*    denom   = (float*)carve((size_t)NN * 4);
    _Float16* hwT16   = (_Float16*)carve((size_t)NT_ * 256 * 256 * 2);
    _Float16* lwT16   = (_Float16*)carve((size_t)288 * 256 * 2);

    for (int l = 0; l < 5; ++l){
        int ci = DIMS_[l], co = DIMS_[l + 1];
        const float* hw  = (const float*)d_in[2 + 9 * l + 0];
        const float* hb  = (const float*)d_in[2 + 9 * l + 1];
        const float* ete = (const float*)d_in[2 + 9 * l + 2];
        const float* eaw = (const float*)d_in[2 + 9 * l + 3];
        const float* att = (const float*)d_in[2 + 9 * l + 4];
        const float* lw  = (const float*)d_in[2 + 9 * l + 5];
        const float* lb  = (const float*)d_in[2 + 9 * l + 6];
        const float* g   = (const float*)d_in[2 + 9 * l + 7];
        const float* b   = (const float*)d_in[2 + 9 * l + 8];
        const float* hin = (l == 0) ? x : h_cur;

        int groups = (co + 127) >> 7;

        { size_t n = (size_t)NN * ci;
          k_f32_to_f16<<<(unsigned)((n + 255) / 256), 256, 0, stream>>>(hin, hf16, n); }
        { size_t n = (size_t)NT_ * ci * co;
          k_transpose_f16<<<(unsigned)((n + 255) / 256), 256, 0, stream>>>(hw, hwT16, ci, co, n); }
        { size_t n = (size_t)(co + 32) * co;
          k_transpose_f16<<<(unsigned)((n + 255) / 256), 256, 0, stream>>>(lw, lwT16, co + 32, co, n); }

        k_hetero_gemm<<<(NN / 16) * groups, 256, 0, stream>>>(hf16, hwT16, hb, node_type, xl, xlf16, ci, co);
        k_node_dots<<<(NN + 255) / 256, 256, 0, stream>>>(xl, att, s_i, s_j, co);
        k_init_node<<<(NN + 255) / 256, 256, 0, stream>>>(menc, denom, NN);
        { size_t n = (size_t)NN * co;
          k_zero<<<(unsigned)((n + 255) / 256), 256, 0, stream>>>(hnxt, n); }
        k_edge_emb<<<(EE + 255) / 256, 256, 0, stream>>>(edge_attr, eaw, att + 2 * co + D_ET_, eaf16, eadot);
        k_alpha<<<(EE + 255) / 256, 256, 0, stream>>>(srcp, dstp, edge_type, s_i, s_j, ete, att + 2 * co, eadot, alphab, menc);
        k_exp<<<(EE + 255) / 256, 256, 0, stream>>>(dstp, alphab, menc, exv, denom);
        k_gemm_bias<<<(NN / 16) * groups, 256, 0, stream>>>(xlf16, lwT16, lb, msgnode, co, co, co + 32);
        k_msg_scatter<<<(EE / 16) * groups, 256, 0, stream>>>(eaf16, lwT16, msgnode, srcp, dstp, exv, denom, hnxt, co);

        float* lnout = (l < 4) ? h_cur : (float*)d_out;
        k_layernorm<<<NN / 8, 256, 0, stream>>>(hnxt, g, b, lnout, co, (l < 4) ? 1 : 0);
    }
}
